// Temporal_Multimodal_Decoder_15144054686171
// MI455X (gfx1250) — compile-verified, matrix operand order checked by reading, not run on hardware
//
#include <hip/hip_runtime.h>
#include <hip/hip_bf16.h>
#include <math.h>

// ---------------------------------------------------------------------------
// Temporal multimodal LSTM decoder for MI455X (gfx1250).
// Batch rows are independent through the recurrence, so each workgroup owns a
// 16-row tile and runs all T=60 steps locally using v_wmma_f32_16x16x32_bf16
// for every GEMM (emb 100->200, gates 480->1024, conf 256->256). Weights are
// repacked once into bf16 [N][K] K-contiguous layout so each B-fragment is two
// contiguous 16B global loads per lane, matching the CDNA5 16-bit matrix VGPR
// layout. LSTM cell state lives in registers (thread t owns hidden unit t),
// activations in LDS (<64KB). Pointwise uses the gfx1250 hardware tanh/exp
// TRANS ops where available.
// ---------------------------------------------------------------------------

#define Bsz 4096
#define WS  50
#define Hd  256
#define Dd  2
#define Mm  6
#define Tt  60

typedef __attribute__((ext_vector_type(16))) __bf16 bf16x16;
typedef __attribute__((ext_vector_type(8)))  float  f32x8;

__device__ __forceinline__ unsigned short f2bf(float f) {
  unsigned int u = __float_as_uint(f);
  unsigned int r = (u + 0x7FFFu + ((u >> 16) & 1u)) >> 16;  // RNE
  return (unsigned short)r;
}

__device__ __forceinline__ float fast_tanh(float x) {
#if __has_builtin(__builtin_amdgcn_tanhf)
  return __builtin_amdgcn_tanhf(x);          // v_tanh_f32 (gfx1250 TRANS op)
#elif __has_builtin(__builtin_amdgcn_tanh_f32)
  return __builtin_amdgcn_tanh_f32(x);
#else
  return tanhf(x);
#endif
}

__device__ __forceinline__ float sigf(float x) { return 1.f / (1.f + __expf(-x)); }

union BFrag { uint4 q[2]; bf16x16 v; };

// Fragment loader for the CDNA5 16-bit A/B layout:
// lanes 0-15: K = k0+0..7 and k0+16..23 ; lanes 16-31: K = k0+8..15, k0+24..31.
// Caller passes p0 = base + (row-or-col)*stride + k0 + (lane>>4)*8.
__device__ __forceinline__ bf16x16 load_frag(const unsigned short* p0) {
  BFrag u;
  u.q[0] = *(const uint4*)(p0);
  u.q[1] = *(const uint4*)(p0 + 16);
  return u.v;
}

__device__ __forceinline__ f32x8 wmma_bf16(bf16x16 a, bf16x16 b, f32x8 c) {
  return __builtin_amdgcn_wmma_f32_16x16x32_bf16(false, a, false, b, (short)0, c,
                                                 false, false);
}

// ---------------------------------------------------------------------------
// Workspace layout (bf16 elements):
//   W_gT   [1024][512] : combined gate weights, row n holds K=0..199 -> W_ih,
//                        224..479 -> W_hh, zero pad elsewhere.
//   W_embT [ 208][128] : embedding weights transposed, zero padded.
//   W1T/W2T/W3T [256][256] : confidence MLP weights transposed.
// ---------------------------------------------------------------------------
#define WGT_OFF   0
#define WEMB_OFF  524288
#define W1T_OFF   550912
#define W2T_OFF   616448
#define W3T_OFF   681984
#define WS_ELEMS  747520

__global__ __launch_bounds__(256) void prep_kernel(
    const float* __restrict__ W_emb, const float* __restrict__ W_ih,
    const float* __restrict__ W_hh,  const float* __restrict__ W1,
    const float* __restrict__ W2,    const float* __restrict__ W3,
    unsigned short* __restrict__ ws) {
  int idx = blockIdx.x * 256 + threadIdx.x;
  if (idx < 524288) {  // W_gT [1024][512]
    int n = idx >> 9, k = idx & 511;
    float v = 0.f;
    if (k < 200)                  v = W_ih[k * 1024 + n];
    else if (k >= 224 && k < 480) v = W_hh[(k - 224) * 1024 + n];
    ws[WGT_OFF + idx] = f2bf(v);
    return;
  }
  idx -= 524288;
  if (idx < 26624) {  // W_embT [208][128]
    int n = idx >> 7, k = idx & 127;
    float v = (n < 200 && k < 100) ? W_emb[k * 200 + n] : 0.f;
    ws[WEMB_OFF + idx] = f2bf(v);
    return;
  }
  idx -= 26624;
  if (idx < 3 * 65536) {  // W1T/W2T/W3T [256][256]
    int l = idx >> 16, r = idx & 65535;
    int n = r >> 8, k = r & 255;
    const float* W = (l == 0) ? W1 : ((l == 1) ? W2 : W3);
    ws[W1T_OFF + idx] = f2bf(W[k * 256 + n]);
  }
}

// ---------------------------------------------------------------------------
// Decoder: grid (B/16, M), 256 threads (8 waves).
// LDS map (57600B):
//   [0      .. 32768) : gates f32 [16][512] (half-unit mapping)
//   [32768  .. 49152) : xh bf16 [16][512]  (K: 0..199 emb-x, 224..479 h)
//   [49152  .. 52480) : traj ring bf16 [16][52][2]
//   [52480  .. 53504) : rel reduction f32 [8][16][2]
//   [53504  .. 57600) : x_in staging bf16 [16][128]
// ---------------------------------------------------------------------------
#define DEC_SMEM 57600

__global__ __launch_bounds__(256) void decoder_kernel(
    const float* __restrict__ traj_in, const float* __restrict__ h0,
    const float* __restrict__ b_emb,   const float* __restrict__ b_ih,
    const float* __restrict__ b_hh,    const float* __restrict__ W_pred,
    const float* __restrict__ b_pred,
    const unsigned short* __restrict__ W_gT,
    const unsigned short* __restrict__ W_embT,
    float* __restrict__ out) {
  __shared__ uint4 smem4[DEC_SMEM / 16];
  unsigned char* smem = (unsigned char*)smem4;
  float*          s_gates = (float*)smem;                       // [16][512]
  unsigned short* s_xh    = (unsigned short*)(smem + 32768);    // [16][512]
  unsigned short* s_traj  = (unsigned short*)(smem + 49152);    // [16][52*2]
  float*          s_red   = (float*)(smem + 52480);             // [8][16][2]
  unsigned short* s_xin   = (unsigned short*)(smem + 53504);    // [16][128]

  const int tid  = threadIdx.x;
  const int lane = tid & 31;
  const int wave = tid >> 5;
  const int b0   = blockIdx.x * 16;
  const int mode = blockIdx.y;

  // ---- init trajectory ring in bf16 (head = 0 -> flat (w,d) order) ----
  for (int idx = tid; idx < 16 * 100; idx += 256) {
    int row = idx / 100, r = idx - row * 100;
    s_traj[row * 104 + r] = f2bf(traj_in[(b0 + row) * 100 + r]);
  }
  // ---- init xh: h in K[224..480), zeros elsewhere (pads stay zero) ----
  for (int idx = tid; idx < 16 * 512; idx += 256) {
    int row = idx >> 9, k = idx & 511;
    float v = (k >= 224 && k < 480) ? h0[(b0 + row) * 256 + (k - 224)] : 0.f;
    s_xh[row * 512 + k] = f2bf(v);
  }
  // ---- zero the x_in K pad [100..128) once (never rewritten) ----
  for (int idx = tid; idx < 2048; idx += 256) {
    if ((idx & 127) >= 100) s_xin[idx] = 0;
  }

  float creg[16], hreg[16];
#pragma unroll
  for (int j = 0; j < 16; ++j) { creg[j] = 0.f; hreg[j] = 0.f; }

  // ---- step-invariant per-lane constants ----
  const float wp0 = W_pred[(mode * 256 + tid) * 2 + 0];  // thread t owns unit t
  const float wp1 = W_pred[(mode * 256 + tid) * 2 + 1];
  const int coln0 = wave * 16 + (lane & 15);             // emb tile t=wave
  const int coln1 = coln0 + 128;                          // emb tile t=wave+8
  const float ebias0 = (coln0 < 200) ? b_emb[coln0] : 0.f;
  const float ebias1 = (coln1 < 200) ? b_emb[coln1] : 0.f;
  float gbias[8];
#pragma unroll
  for (int t = 0; t < 8; ++t) {
    int n = wave * 128 + t * 16 + (lane & 15);
    gbias[t] = b_ih[n] + b_hh[n];
  }
  int head = 0;
  __syncthreads();

  for (int step = 0; step < Tt; ++step) {
    // ---- 1. stage x_in bf16 [16][100] from the bf16 ring (plain copies) --
    for (int idx = tid; idx < 1600; idx += 256) {
      int row = idx / 100, k = idx - row * 100;
      int slot = head + (k >> 1); if (slot >= 50) slot -= 50;
      s_xin[row * 128 + k] = s_traj[row * 104 + slot * 2 + (k & 1)];
    }
    __syncthreads();

    // ---- 2. embedding GEMM (16x100 @ 100x200) + leaky_relu -> xh[0..200) --
    {
      const int row = lane & 15, kh = lane >> 4;
      for (int t = wave; t < 13; t += 8) {
        const int coln = t * 16 + (lane & 15);
        f32x8 acc = {0.f, 0.f, 0.f, 0.f, 0.f, 0.f, 0.f, 0.f};
#pragma unroll
        for (int kk = 0; kk < 4; ++kk) {
          bf16x16 a = load_frag(s_xin + row * 128 + kk * 32 + kh * 8);
          bf16x16 b = load_frag(W_embT + coln * 128 + kk * 32 + kh * 8);
          acc = wmma_bf16(a, b, acc);
        }
        if (coln < 200) {
          float bias = (t == wave) ? ebias0 : ebias1;
#pragma unroll
          for (int r = 0; r < 8; ++r) {
            float v = acc[r] + bias;
            v = (v > 0.f) ? v : 0.01f * v;  // leaky_relu, slope 0.01
            s_xh[(r + 8 * kh) * 512 + coln] = f2bf(v);
          }
        }
      }
    }
    __syncthreads();

    // ---- 3. gates GEMM: (16x480) @ (480x1024), each wave 8 N-tiles -------
    f32x8 acc[8];
#pragma unroll
    for (int t = 0; t < 8; ++t)
      acc[t] = (f32x8){0.f, 0.f, 0.f, 0.f, 0.f, 0.f, 0.f, 0.f};
    {
      const int row = lane & 15, kh = lane >> 4;
      const int colbase = wave * 128 + (lane & 15);
      for (int kk = 0; kk < 15; ++kk) {
        bf16x16 a = load_frag(s_xh + row * 512 + kk * 32 + kh * 8);
#pragma unroll
        for (int t = 0; t < 8; ++t) {
          bf16x16 b = load_frag(W_gT + (colbase + t * 16) * 512 + kk * 32 + kh * 8);
          acc[t] = wmma_bf16(a, b, acc[t]);
        }
      }
    }
    __syncthreads();  // xh fully consumed; safe to rewrite gates + xh h-region

    // ---- 4. two half-unit passes: writeback + LSTM pointwise -------------
    // cmap packs gate column n -> (n>>8)*128 + (n&127); pass p holds units
    // u with (u>>7)==p, i.e. tiles of waves with matching parity.
#pragma unroll
    for (int p = 0; p < 2; ++p) {
      if ((wave & 1) == p) {
        const int kh = lane >> 4;
#pragma unroll
        for (int t = 0; t < 8; ++t) {
          int n = wave * 128 + t * 16 + (lane & 15);
          int cmap = ((n >> 8) << 7) + (n & 127);
#pragma unroll
          for (int r = 0; r < 8; ++r)
            s_gates[(r + 8 * kh) * 512 + cmap] = acc[t][r] + gbias[t];
        }
      }
      __syncthreads();
      if ((tid >> 7) == p) {
        const int uc = tid & 127;
#pragma unroll
        for (int j = 0; j < 16; ++j) {
          float gi = s_gates[j * 512 + 0 * 128 + uc];
          float gf = s_gates[j * 512 + 1 * 128 + uc];
          float gg = s_gates[j * 512 + 2 * 128 + uc];
          float go = s_gates[j * 512 + 3 * 128 + uc];
          float c = sigf(gf) * creg[j] + sigf(gi) * fast_tanh(gg);
          float h = sigf(go) * fast_tanh(c);
          creg[j] = c;
          hreg[j] = h;
          s_xh[j * 512 + 224 + tid] = f2bf(h);  // h feeds next step's GEMM
        }
      }
      __syncthreads();
    }

    // ---- 5. rel = h @ Wp + bp via wave shuffle reduction ----------------
#pragma unroll
    for (int j = 0; j < 16; ++j) {
      float a = hreg[j] * wp0, b = hreg[j] * wp1;
#pragma unroll
      for (int off = 16; off > 0; off >>= 1) {
        a += __shfl_xor(a, off, 32);
        b += __shfl_xor(b, off, 32);
      }
      if (lane == 0) {
        s_red[(wave * 16 + j) * 2 + 0] = a;
        s_red[(wave * 16 + j) * 2 + 1] = b;
      }
    }
    __syncthreads();
    if (tid < 32) {
      int j = tid >> 1, d = tid & 1;
      float s = b_pred[mode * 2 + d];
#pragma unroll
      for (int w = 0; w < 8; ++w) s += s_red[(w * 16 + j) * 2 + d];
      out[(((b0 + j) * Mm + mode) * Tt + step) * 2 + d] = s;
      s_traj[j * 104 + head * 2 + d] = f2bf(s);  // append into ring at old head
    }
    __syncthreads();
    head = head + 1; if (head >= 50) head = 0;
  }
}

// ---------------------------------------------------------------------------
// Confidence head: 3x LinearRes(H->H with eval BN) + Linear(H->M) + softmax.
// ---------------------------------------------------------------------------
#define CONF_SMEM 41472

__global__ __launch_bounds__(256) void conf_kernel(
    const float* __restrict__ h0,
    const float* __restrict__ b1, const float* __restrict__ b2, const float* __restrict__ b3,
    const float* __restrict__ g1, const float* __restrict__ be1, const float* __restrict__ m1, const float* __restrict__ v1,
    const float* __restrict__ g2, const float* __restrict__ be2, const float* __restrict__ m2, const float* __restrict__ v2,
    const float* __restrict__ g3, const float* __restrict__ be3, const float* __restrict__ m3, const float* __restrict__ v3,
    const float* __restrict__ Wc, const float* __restrict__ bc,
    const unsigned short* __restrict__ W1T, const unsigned short* __restrict__ W2T,
    const unsigned short* __restrict__ W3T, float* __restrict__ conf_out) {
  __shared__ uint4 smem4[CONF_SMEM / 16];
  unsigned char* smem = (unsigned char*)smem4;
  unsigned short* s_xb = (unsigned short*)smem;        // [16][256] bf16
  float* s_x0   = (float*)(smem + 8192);               // [16][256] residual
  float* s_y    = (float*)(smem + 8192 + 16384);       // [16][256] layer-3 out
  float* s_conf = (float*)(smem + 8192 + 32768);       // [16][8]

  const int tid = threadIdx.x, lane = tid & 31, wave = tid >> 5;
  const int b0 = blockIdx.x * 16;

  for (int idx = tid; idx < 4096; idx += 256) {
    int row = idx >> 8;
    float v = h0[(b0 + row) * 256 + (idx & 255)];
    s_x0[idx] = v;
    s_xb[idx] = f2bf(v);
  }
  __syncthreads();

  for (int layer = 0; layer < 3; ++layer) {
    const unsigned short* Wt = (layer == 0) ? W1T : ((layer == 1) ? W2T : W3T);
    const float* bb = (layer == 0) ? b1 : ((layer == 1) ? b2 : b3);
    const float* gg = (layer == 0) ? g1 : ((layer == 1) ? g2 : g3);
    const float* be = (layer == 0) ? be1 : ((layer == 1) ? be2 : be3);
    const float* mm = (layer == 0) ? m1 : ((layer == 1) ? m2 : m3);
    const float* vv = (layer == 0) ? v1 : ((layer == 1) ? v2 : v3);
    f32x8 acc[2];
    acc[0] = (f32x8){0.f, 0.f, 0.f, 0.f, 0.f, 0.f, 0.f, 0.f};
    acc[1] = (f32x8){0.f, 0.f, 0.f, 0.f, 0.f, 0.f, 0.f, 0.f};
    const int row = lane & 15, kh = lane >> 4;
    for (int kk = 0; kk < 8; ++kk) {
      bf16x16 a = load_frag(s_xb + row * 256 + kk * 32 + kh * 8);
#pragma unroll
      for (int t = 0; t < 2; ++t) {
        int n0 = (wave * 2 + t) * 16;
        bf16x16 b = load_frag(Wt + (n0 + (lane & 15)) * 256 + kk * 32 + kh * 8);
        acc[t] = wmma_bf16(a, b, acc[t]);
      }
    }
    __syncthreads();  // all reads of s_xb done
#pragma unroll
    for (int t = 0; t < 2; ++t) {
      int n = (wave * 2 + t) * 16 + (lane & 15);
      float scale = gg[n] * rsqrtf(vv[n] + 1e-5f);
      float shift = be[n] - mm[n] * scale;
#pragma unroll
      for (int r = 0; r < 8; ++r) {
        int rrow = r + 8 * kh;
        float y = (acc[t][r] + bb[n]) * scale + shift;  // eval-mode BN
        if (layer < 2) {
          y = fmaxf(y, 0.f);
          s_xb[rrow * 256 + n] = f2bf(y);
        } else {
          s_y[rrow * 256 + n] = y + s_x0[rrow * 256 + n];  // identity shortcut
        }
      }
    }
    __syncthreads();
  }
  // head + softmax (relu applied inline)
  if (tid < 96) {
    int row = tid / 6, mmod = tid % 6;
    float s = bc[mmod];
    for (int u = 0; u < 256; ++u)
      s += fmaxf(s_y[row * 256 + u], 0.f) * Wc[u * 6 + mmod];
    s_conf[row * 8 + mmod] = s;
  }
  __syncthreads();
  if (tid < 16) {
    float mx = -1e30f;
    for (int k = 0; k < 6; ++k) mx = fmaxf(mx, s_conf[tid * 8 + k]);
    float sum = 0.f, e[6];
    for (int k = 0; k < 6; ++k) { e[k] = __expf(s_conf[tid * 8 + k] - mx); sum += e[k]; }
    for (int k = 0; k < 6; ++k) conf_out[(b0 + tid) * 6 + k] = e[k] / sum;
  }
}

// ---------------------------------------------------------------------------
extern "C" void kernel_launch(void* const* d_in, const int* in_sizes, int n_in,
                              void* d_out, int out_size, void* d_ws, size_t ws_size,
                              hipStream_t stream) {
  // setup_inputs() order: traj_rel, state_tuple_h, then params in insertion
  // order: W_emb,b_emb,W_ih,W_hh,b_ih,b_hh,W_pred,b_pred,W1,b1,W2,b2,W3,b3,
  //        g1,be1,m1,v1,g2,be2,m2,v2,g3,be3,m3,v3,Wc,bc
  const float* traj   = (const float*)d_in[0];
  const float* h0     = (const float*)d_in[1];
  const float* W_emb  = (const float*)d_in[2];
  const float* b_emb  = (const float*)d_in[3];
  const float* W_ih   = (const float*)d_in[4];
  const float* W_hh   = (const float*)d_in[5];
  const float* b_ih   = (const float*)d_in[6];
  const float* b_hh   = (const float*)d_in[7];
  const float* W_pred = (const float*)d_in[8];
  const float* b_pred = (const float*)d_in[9];
  const float* W1 = (const float*)d_in[10]; const float* b1 = (const float*)d_in[11];
  const float* W2 = (const float*)d_in[12]; const float* b2 = (const float*)d_in[13];
  const float* W3 = (const float*)d_in[14]; const float* b3 = (const float*)d_in[15];
  const float* g1 = (const float*)d_in[16]; const float* be1 = (const float*)d_in[17];
  const float* m1 = (const float*)d_in[18]; const float* v1  = (const float*)d_in[19];
  const float* g2 = (const float*)d_in[20]; const float* be2 = (const float*)d_in[21];
  const float* m2 = (const float*)d_in[22]; const float* v2  = (const float*)d_in[23];
  const float* g3 = (const float*)d_in[24]; const float* be3 = (const float*)d_in[25];
  const float* m3 = (const float*)d_in[26]; const float* v3  = (const float*)d_in[27];
  const float* Wc = (const float*)d_in[28]; const float* bc  = (const float*)d_in[29];

  unsigned short* ws = (unsigned short*)d_ws;
  float* out = (float*)d_out;
  float* conf_out = out + (size_t)Bsz * Mm * Tt * Dd;

  const int prep_blocks = (WS_ELEMS + 255) / 256;
  prep_kernel<<<prep_blocks, 256, 0, stream>>>(W_emb, W_ih, W_hh, W1, W2, W3, ws);

  dim3 dgrid(Bsz / 16, Mm);
  decoder_kernel<<<dgrid, 256, 0, stream>>>(
      traj, h0, b_emb, b_ih, b_hh, W_pred, b_pred,
      ws + WGT_OFF, ws + WEMB_OFF, out);

  conf_kernel<<<Bsz / 16, 256, 0, stream>>>(
      h0, b1, b2, b3,
      g1, be1, m1, v1, g2, be2, m2, v2, g3, be3, m3, v3,
      Wc, bc, ws + W1T_OFF, ws + W2T_OFF, ws + W3T_OFF, conf_out);
}
